// SelfAttn_9234179686471
// MI455X (gfx1250) — matrix operand order repaired and checked by reading, not compiled, not run on hardware
//
#include <hip/hip_runtime.h>
#include <hip/hip_bf16.h>

#define NSESS 32
#define EDIM  1024
#define NHEAD 16
#define HDIM  64
#define QKVDIM (3 * EDIM)

typedef __bf16 bf16_t;
typedef __attribute__((ext_vector_type(16))) __bf16 v16bf;
typedef __attribute__((ext_vector_type(8)))  float  v8f;

static __device__ __forceinline__ v8f wmma_bf16f32(v16bf a, v16bf b, v8f c) {
  // D = A(16x32 bf16) * B(32x16 bf16) + C(16x16 f32)
  return __builtin_amdgcn_wmma_f32_16x16x32_bf16(false, a, false, b, (short)0, c,
                                                 false, false);
}

// A fragment: 16(M) x 32(K) bf16 tile at 'base' (row-major, leading dim lda)
// lane m = lane&15, half = lane>>4; element e -> k = 16*half + e
static __device__ __forceinline__ v16bf frag_a(const bf16_t* base, int lda) {
  const int lane = threadIdx.x & 31;
  const int m = lane & 15;
  const int half = lane >> 4;
  const bf16_t* r = base + m * lda;
  v16bf a;
#pragma unroll
  for (int e = 0; e < 16; ++e) a[e] = r[16 * half + e];
  return a;
}

// B fragment (32K x 16N) where B[k][n] = W[n][k], W row-major with leading dim ldw.
// 'w' points at W[nTile][kTile]. b[e] = W[nTile+n][kTile + 16*half + e] (contiguous).
static __device__ __forceinline__ v16bf frag_b_trans(const bf16_t* w, int ldw) {
  const int lane = threadIdx.x & 31;
  const int n = lane & 15;
  const int half = lane >> 4;
  const bf16_t* r = w + n * ldw + 16 * half;
  v16bf b;
#pragma unroll
  for (int e = 0; e < 16; ++e) b[e] = r[e];
  return b;
}

// B fragment from row-major B (K rows x N cols, leading dim ldb): b[e] = B[16*half+e][n]
static __device__ __forceinline__ v16bf frag_b_rowmajor(const bf16_t* bm, int ldb) {
  const int lane = threadIdx.x & 31;
  const int n = lane & 15;
  const int half = lane >> 4;
  const bf16_t* r = bm + (16 * half) * ldb + n;
  v16bf b;
#pragma unroll
  for (int e = 0; e < 16; ++e) b[e] = r[e * ldb];
  return b;
}

// ---------------- prep kernels ----------------

__global__ void k_offsets(const int* __restrict__ sections, int* __restrict__ off) {
  if (threadIdx.x == 0 && blockIdx.x == 0) {
    int s = 0;
    for (int b = 0; b < NSESS; ++b) { off[b] = s; s += sections[b]; }
    off[NSESS] = s;
  }
}

__global__ void k_cvt_bf16(const float* __restrict__ x, bf16_t* __restrict__ y, int n) {
  int i = blockIdx.x * blockDim.x + threadIdx.x;
  if (i < n) y[i] = (bf16_t)x[i];
}

// Pad/gather sessions into Xbf [B, Lpad, E] (bf16), zeros beyond session length.
__global__ void k_pad(const float* __restrict__ sess, const int* __restrict__ sections,
                      const int* __restrict__ off, bf16_t* __restrict__ xbf, int Lpad) {
  int i = blockIdx.x * blockDim.x + threadIdx.x;
  int total = NSESS * Lpad * EDIM;
  if (i >= total) return;
  int k = i & (EDIM - 1);
  int l = (i / EDIM) % Lpad;
  int b = i / (EDIM * Lpad);
  float v = 0.0f;
  if (l < sections[b]) v = sess[(off[b] + l) * EDIM + k];
  xbf[i] = (bf16_t)v;
}

// ---------------- QKV projection: qkv = Xpad @ in_proj_w^T + in_proj_b ----------------
// grid: (Lpad/16, 3072/64, B), block 32 (one wave computes 16x64 outputs)
__global__ __launch_bounds__(32)
void k_qkv(const bf16_t* __restrict__ Xbf, const bf16_t* __restrict__ Wbf,
           const float* __restrict__ bias,
           bf16_t* __restrict__ Qbf, bf16_t* __restrict__ Kbf, bf16_t* __restrict__ Vbf,
           int Lpad) {
  const int mt = blockIdx.x;
  const int ns = blockIdx.y;
  const int b  = blockIdx.z;
  const int lane = threadIdx.x & 31;
  const int nlo = lane & 15, half = lane >> 4;

  const bf16_t* A0 = Xbf + (b * Lpad + mt * 16) * EDIM;
  const int n0 = ns * 64;

  v8f acc[4];
#pragma unroll
  for (int t = 0; t < 4; ++t)
#pragma unroll
    for (int e = 0; e < 8; ++e) acc[t][e] = 0.0f;

  for (int k0 = 0; k0 < EDIM; k0 += 32) {
    v16bf a = frag_a(A0 + k0, EDIM);
#pragma unroll
    for (int t = 0; t < 4; ++t) {
      v16bf bfrag = frag_b_trans(Wbf + (n0 + t * 16) * EDIM + k0, EDIM);
      acc[t] = wmma_bf16f32(a, bfrag, acc[t]);
    }
  }

#pragma unroll
  for (int t = 0; t < 4; ++t) {
    const int n = n0 + t * 16 + nlo;
    const float bv = bias[n];
    const int which = n >> 10;      // 0:Q 1:K 2:V
    const int nn = n & 1023;
    const int h = nn >> 6;
    const int d = nn & 63;
    bf16_t* dst = (which == 0) ? Qbf : (which == 1) ? Kbf : Vbf;
    bf16_t* p = dst + ((b * NHEAD + h) * Lpad) * HDIM + d;
#pragma unroll
    for (int e = 0; e < 8; ++e) {
      int row = mt * 16 + e + 8 * half;
      p[row * HDIM] = (bf16_t)(acc[t][e] + bv);
    }
  }
}

// ---------------- flash attention per (b, h, 16-query tile) ----------------
__global__ __launch_bounds__(32)
void k_attn(const bf16_t* __restrict__ Qbf, const bf16_t* __restrict__ Kbf,
            const bf16_t* __restrict__ Vbf, bf16_t* __restrict__ Obf,
            int Lpad, int L) {
  const int mt = blockIdx.x;
  const int h  = blockIdx.y;
  const int b  = blockIdx.z;
  const int lane = threadIdx.x & 31;
  const int nlo = lane & 15, half = lane >> 4;

  const int head = b * NHEAD + h;
  const bf16_t* Qh = Qbf + head * Lpad * HDIM;
  const bf16_t* Kh = Kbf + head * Lpad * HDIM;
  const bf16_t* Vh = Vbf + head * Lpad * HDIM;

  __shared__ bf16_t Pl[16 * 32];

  v16bf qa0 = frag_a(Qh + (mt * 16) * HDIM + 0,  HDIM);
  v16bf qa1 = frag_a(Qh + (mt * 16) * HDIM + 32, HDIM);

  v8f oacc[4];
  float mrow[8], lrow[8];
#pragma unroll
  for (int t = 0; t < 4; ++t)
#pragma unroll
    for (int e = 0; e < 8; ++e) oacc[t][e] = 0.0f;
#pragma unroll
  for (int e = 0; e < 8; ++e) { mrow[e] = -3.0e38f; lrow[e] = 0.0f; }

  const float scale = 0.125f;  // 1/sqrt(HD=64)

  for (int j0 = 0; j0 < Lpad; j0 += 32) {
    v8f s0, s1;
#pragma unroll
    for (int e = 0; e < 8; ++e) { s0[e] = 0.0f; s1[e] = 0.0f; }

    // scores: S[m, j0+n] = sum_k Q[m,k] * K[j0+n, k]
    v16bf kb;
    kb = frag_b_trans(Kh + j0 * HDIM + 0,  HDIM); s0 = wmma_bf16f32(qa0, kb, s0);
    kb = frag_b_trans(Kh + j0 * HDIM + 32, HDIM); s0 = wmma_bf16f32(qa1, kb, s0);
    kb = frag_b_trans(Kh + (j0 + 16) * HDIM + 0,  HDIM); s1 = wmma_bf16f32(qa0, kb, s1);
    kb = frag_b_trans(Kh + (j0 + 16) * HDIM + 32, HDIM); s1 = wmma_bf16f32(qa1, kb, s1);

    const bool mask0 = (j0 + nlo) >= L;
    const bool mask1 = (j0 + 16 + nlo) >= L;
#pragma unroll
    for (int e = 0; e < 8; ++e) {
      float v0 = mask0 ? -3.0e38f : s0[e] * scale;
      float v1 = mask1 ? -3.0e38f : s1[e] * scale;
      // row max across the 16 lanes of this half (columns 0..15 of both tiles)
      float tm = fmaxf(v0, v1);
      tm = fmaxf(tm, __shfl_xor(tm, 1, 32));
      tm = fmaxf(tm, __shfl_xor(tm, 2, 32));
      tm = fmaxf(tm, __shfl_xor(tm, 4, 32));
      tm = fmaxf(tm, __shfl_xor(tm, 8, 32));
      float nm = fmaxf(mrow[e], tm);
      float corr = __expf(mrow[e] - nm);
      float p0 = __expf(v0 - nm);
      float p1 = __expf(v1 - nm);
      float ts = p0 + p1;
      ts += __shfl_xor(ts, 1, 32);
      ts += __shfl_xor(ts, 2, 32);
      ts += __shfl_xor(ts, 4, 32);
      ts += __shfl_xor(ts, 8, 32);
      lrow[e] = lrow[e] * corr + ts;
      mrow[e] = nm;
#pragma unroll
      for (int t = 0; t < 4; ++t) oacc[t][e] *= corr;
      // stage P (C-fragment layout) into LDS so it can be reloaded in A layout
      Pl[(e + 8 * half) * 32 + nlo]      = (bf16_t)p0;
      Pl[(e + 8 * half) * 32 + nlo + 16] = (bf16_t)p1;
    }
    __syncthreads();
    v16bf pa = frag_a(Pl, 32);          // P as 16x32 A fragment (K = 32 keys)
#pragma unroll
    for (int t = 0; t < 4; ++t) {
      v16bf vb = frag_b_rowmajor(Vh + j0 * HDIM + t * 16, HDIM);
      oacc[t] = wmma_bf16f32(pa, vb, oacc[t]);
    }
    __syncthreads();
  }

  bf16_t* Ob = Obf + (b * Lpad) * EDIM + h * HDIM;
#pragma unroll
  for (int t = 0; t < 4; ++t) {
#pragma unroll
    for (int e = 0; e < 8; ++e) {
      int row = mt * 16 + e + 8 * half;
      if (row < L) Ob[row * EDIM + t * 16 + nlo] = (bf16_t)(oacc[t][e] / lrow[e]);
    }
  }
}

// ---------------- output projection: out = O @ out_proj_w^T + out_proj_b ----------------
__global__ __launch_bounds__(32)
void k_outproj(const bf16_t* __restrict__ Obf, const bf16_t* __restrict__ Wbf,
               const float* __restrict__ bias, float* __restrict__ out,
               int Lpad, int L) {
  const int mt = blockIdx.x;
  const int ns = blockIdx.y;
  const int b  = blockIdx.z;
  const int lane = threadIdx.x & 31;
  const int nlo = lane & 15, half = lane >> 4;

  const bf16_t* A0 = Obf + (b * Lpad + mt * 16) * EDIM;
  const int n0 = ns * 64;

  v8f acc[4];
#pragma unroll
  for (int t = 0; t < 4; ++t)
#pragma unroll
    for (int e = 0; e < 8; ++e) acc[t][e] = 0.0f;

  for (int k0 = 0; k0 < EDIM; k0 += 32) {
    v16bf a = frag_a(A0 + k0, EDIM);
#pragma unroll
    for (int t = 0; t < 4; ++t) {
      v16bf bfrag = frag_b_trans(Wbf + (n0 + t * 16) * EDIM + k0, EDIM);
      acc[t] = wmma_bf16f32(a, bfrag, acc[t]);
    }
  }

#pragma unroll
  for (int t = 0; t < 4; ++t) {
    const int n = n0 + t * 16 + nlo;
    const float bv = bias[n];
#pragma unroll
    for (int e = 0; e < 8; ++e) {
      int row = mt * 16 + e + 8 * half;
      if (row < L) out[(b * L + row) * EDIM + n] = acc[t][e] + bv;
    }
  }
}

extern "C" void kernel_launch(void* const* d_in, const int* in_sizes, int n_in,
                              void* d_out, int out_size, void* d_ws, size_t ws_size,
                              hipStream_t stream) {
  (void)in_sizes; (void)n_in; (void)ws_size;
  const float* sess       = (const float*)d_in[0];
  const float* in_proj_w  = (const float*)d_in[1];
  const float* in_proj_b  = (const float*)d_in[2];
  const float* out_proj_w = (const float*)d_in[3];
  const float* out_proj_b = (const float*)d_in[4];
  const int*   sections   = (const int*)d_in[5];

  const int L    = out_size / (NSESS * EDIM);   // padded length from output shape
  const int Lpad = (L + 31) & ~31;

  char* ws = (char*)d_ws;
  size_t cur = 0;
  auto carve = [&](size_t bytes) -> void* {
    void* p = ws + cur;
    cur += (bytes + 255) & ~(size_t)255;
    return p;
  };
  int*    off  = (int*)carve(64 * sizeof(int));
  bf16_t* Wqkv = (bf16_t*)carve((size_t)QKVDIM * EDIM * sizeof(bf16_t));
  bf16_t* Wout = (bf16_t*)carve((size_t)EDIM * EDIM * sizeof(bf16_t));
  const size_t nPad = (size_t)NSESS * Lpad * EDIM;
  bf16_t* Xbf = (bf16_t*)carve(nPad * sizeof(bf16_t));
  bf16_t* Qbf = (bf16_t*)carve(nPad * sizeof(bf16_t));
  bf16_t* Kbf = (bf16_t*)carve(nPad * sizeof(bf16_t));
  bf16_t* Vbf = (bf16_t*)carve(nPad * sizeof(bf16_t));
  bf16_t* Obf = (bf16_t*)carve(nPad * sizeof(bf16_t));

  k_offsets<<<1, 32, 0, stream>>>(sections, off);
  {
    int n = QKVDIM * EDIM;
    k_cvt_bf16<<<(n + 255) / 256, 256, 0, stream>>>(in_proj_w, Wqkv, n);
  }
  {
    int n = EDIM * EDIM;
    k_cvt_bf16<<<(n + 255) / 256, 256, 0, stream>>>(out_proj_w, Wout, n);
  }
  {
    int n = NSESS * Lpad * EDIM;
    k_pad<<<(n + 255) / 256, 256, 0, stream>>>(sess, sections, off, Xbf, Lpad);
  }
  k_qkv<<<dim3(Lpad / 16, QKVDIM / 64, NSESS), 32, 0, stream>>>(
      Xbf, Wqkv, in_proj_b, Qbf, Kbf, Vbf, Lpad);
  k_attn<<<dim3(Lpad / 16, NHEAD, NSESS), 32, 0, stream>>>(
      Qbf, Kbf, Vbf, Obf, Lpad, L);
  k_outproj<<<dim3(Lpad / 16, EDIM / 64, NSESS), 32, 0, stream>>>(
      Obf, Wout, out_proj_b, (float*)d_out, Lpad, L);
}
